// HardBinaryConv_5239860101349
// MI455X (gfx1250) — compile-verified, matrix operand order checked by reading, not run on hardware
//
#include <hip/hip_runtime.h>
#include <hip/hip_bf16.h>

typedef __attribute__((ext_vector_type(16))) _Float16 v16h;
typedef __attribute__((ext_vector_type(8)))  _Float16 v8h;
typedef __attribute__((ext_vector_type(8)))  float    v8f;

union Frag16 { v16h v; v8h h[2]; };

#define NIMG 32
#define CIN  256
#define HH   56
#define WW   56
#define OUTC 256
#define HWSZ (HH*WW)              // 3136 (multiple of 16 -> 16-lane runs never straddle images)
#define MTOT (NIMG*HWSZ)          // 100352
#define KPP  9                    // 3x3 positions
#define OT   128                  // block tile over output channels (GEMM M)
#define MT   128                  // block tile over spatial m      (GEMM N)
#define KT   32                   // WMMA K step
#define NIT  (KPP * (CIN / KT))   // 72 K-steps
#define LDSW 40                   // padded K stride in f16 (80B, 16B-aligned)

// ---------------- scale[o] = mean |w[o,:,:,:]| ----------------
__global__ void hbc_scale_kernel(const float* __restrict__ w, float* __restrict__ scale) {
    __shared__ float red[256];
    const int o = blockIdx.x;
    const float* wo = w + o * (CIN * KPP);
    float s = 0.f;
    for (int i = threadIdx.x; i < CIN * KPP; i += 256) s += fabsf(wo[i]);
    red[threadIdx.x] = s;
    __syncthreads();
    for (int st = 128; st > 0; st >>= 1) {
        if (threadIdx.x < st) red[threadIdx.x] += red[threadIdx.x + st];
        __syncthreads();
    }
    if (threadIdx.x == 0) scale[o] = red[0] * (1.0f / (CIN * KPP));
}

// ---------------- wpack[p][o][c] = (f16) sign(w[o][c][p]) ----------------
__global__ void hbc_pack_kernel(const float* __restrict__ w, _Float16* __restrict__ wpack) {
    const int idx = blockIdx.x * 256 + threadIdx.x;   // (p*256 + o)*256 + c
    const int c = idx & 255;
    const int o = (idx >> 8) & 255;
    const int p = idx >> 16;
    const float v = w[(o * CIN + c) * KPP + p];
    const float s = (v > 0.f) ? 1.f : ((v < 0.f) ? -1.f : 0.f);
    wpack[idx] = (_Float16)s;
}

// ---------------- implicit-GEMM WMMA conv: D[o][m] = sum_k W[o][k] X[k][m] ----------------
__global__ __launch_bounds__(256)
void hbc_gemm_kernel(const float* __restrict__ x, const _Float16* __restrict__ wpack,
                     const float* __restrict__ scale, float* __restrict__ out) {
    // double-buffered tiles: X (B-operand) as [m][k], W (A-operand) as [o][k]
    __shared__ __align__(16) _Float16 Xlds[2][MT * LDSW];
    __shared__ __align__(16) _Float16 Wlds[2][OT * LDSW];
    __shared__ float scale_lds[OT];

    const int t      = threadIdx.x;
    const int m_base = blockIdx.x * MT;
    const int o_base = blockIdx.y * OT;

    if (t < OT) scale_lds[t] = scale[o_base + t];

    // ---- X-load geometry: thread owns (m = m_base + (t&127), k in [xci0, xci0+16)) ----
    const int mi   = t & 127;
    const int xci0 = (t >> 7) * 16;          // 0 or 16
    const int m    = m_base + mi;
    const int img  = m / HWSZ;
    const int hw   = m - img * HWSZ;
    const int oh   = hw / WW;
    const int ow   = hw - oh * WW;

    // ---- W-load geometry: thread owns (o = o_base + (t&127), k in [wci0, wci0+16)) ----
    const int wi   = t & 127;
    const int wci0 = (t >> 7) * 16;

    // ---- wave / lane geometry: 8 waves as 2(O) x 4(m); wave tile = 64(O) x 32(m) ----
    const int wave   = t >> 5;
    const int lane   = t & 31;
    const int wave_o = wave & 1;
    const int wave_m = wave >> 1;
    const int g      = lane >> 4;
    const int r16    = lane & 15;

    v8f acc[4][2];
    #pragma unroll
    for (int i = 0; i < 4; ++i)
        #pragma unroll
        for (int j = 0; j < 2; ++j) acc[i][j] = (v8f){};

    // staging registers for the software pipeline
    float fx[16];
    v8h   wr0, wr1;

    auto load_global = [&](int it) {
        const int p  = it >> 3;
        const int cb = it & 7;
        const int c0 = cb * KT;
        const int kh = p / 3, kw = p - kh * 3;
        const int ih = oh + kh - 1;
        const int iw = ow + kw - 1;
        const bool valid = ((unsigned)ih < (unsigned)HH) && ((unsigned)iw < (unsigned)WW);
        if (valid) {
            const float* xb = x + ((size_t)(img * CIN) * HH + ih) * WW + iw;
            #pragma unroll
            for (int j = 0; j < 16; ++j)
                fx[j] = xb[(size_t)(c0 + xci0 + j) * HWSZ];
            if (cb + 1 < CIN / KT)
                __builtin_prefetch(xb + (size_t)(c0 + KT + xci0) * HWSZ, 0, 0);
        } else {
            #pragma unroll
            for (int j = 0; j < 16; ++j) fx[j] = 0.f;
        }
        const v8h* wsrc = (const v8h*)(wpack + ((size_t)(p * OUTC) + o_base + wi) * CIN
                                             + c0 + wci0);
        wr0 = wsrc[0];
        wr1 = wsrc[1];
    };

    auto store_lds = [&](int buf) {
        v8h h0, h1;
        #pragma unroll
        for (int j = 0; j < 8; ++j) {
            h0[j] = (_Float16)fx[j];
            h1[j] = (_Float16)fx[8 + j];
        }
        *(v8h*)&Xlds[buf][mi * LDSW + xci0]     = h0;
        *(v8h*)&Xlds[buf][mi * LDSW + xci0 + 8] = h1;
        *(v8h*)&Wlds[buf][wi * LDSW + wci0]     = wr0;
        *(v8h*)&Wlds[buf][wi * LDSW + wci0 + 8] = wr1;
    };

    // ---- prologue ----
    load_global(0);
    store_lds(0);
    __syncthreads();

    // ---- main pipeline: one barrier per K-step ----
    #pragma unroll 1
    for (int it = 0; it < NIT; ++it) {
        const int buf = it & 1;
        const bool more = (it + 1) < NIT;
        if (more) load_global(it + 1);        // next tile's global loads in flight

        // fragments per documented 16-bit layout: VGPRs 0..3 = K in [g*8, g*8+8), 4..7 = +16
        Frag16 fw[4], fb[2];
        #pragma unroll
        for (int tm = 0; tm < 4; ++tm) {
            const int row = wave_o * 64 + tm * 16 + r16;
            fw[tm].h[0] = *(const v8h*)&Wlds[buf][row * LDSW + g * 8];
            fw[tm].h[1] = *(const v8h*)&Wlds[buf][row * LDSW + 16 + g * 8];
        }
        #pragma unroll
        for (int tn = 0; tn < 2; ++tn) {
            const int col = wave_m * 32 + tn * 16 + r16;
            fb[tn].h[0] = *(const v8h*)&Xlds[buf][col * LDSW + g * 8];
            fb[tn].h[1] = *(const v8h*)&Xlds[buf][col * LDSW + 16 + g * 8];
        }
        #pragma unroll
        for (int tm = 0; tm < 4; ++tm)
            #pragma unroll
            for (int tn = 0; tn < 2; ++tn)
                acc[tm][tn] = __builtin_amdgcn_wmma_f32_16x16x32_f16(
                    false, fw[tm].v, false, fb[tn].v,
                    (short)0, acc[tm][tn], false, false);

        if (more) {
            store_lds(buf ^ 1);
            __syncthreads();
        }
    }

    // ---- epilogue: D rows = O (uniform per VGPR), cols = spatial m (per lane, contiguous) ----
    #pragma unroll
    for (int tm = 0; tm < 4; ++tm) {
        #pragma unroll
        for (int tn = 0; tn < 2; ++tn) {
            const int mm  = m_base + wave_m * 32 + tn * 16 + r16;   // 16 consecutive per half-wave
            const int im2 = mm / HWSZ;
            const int hw2 = mm - im2 * HWSZ;
            #pragma unroll
            for (int rr = 0; rr < 8; ++rr) {
                const int orow = o_base + wave_o * 64 + tm * 16 + rr + 8 * g;
                const float s  = scale_lds[orow - o_base];
                out[((size_t)(im2 * OUTC + orow)) * HWSZ + hw2] = acc[tm][tn][rr] * s;
            }
        }
    }
}

extern "C" void kernel_launch(void* const* d_in, const int* in_sizes, int n_in,
                              void* d_out, int out_size, void* d_ws, size_t ws_size,
                              hipStream_t stream) {
    (void)in_sizes; (void)n_in; (void)out_size; (void)ws_size;
    const float* x = (const float*)d_in[0];
    const float* w = (const float*)d_in[1];
    float* out = (float*)d_out;

    float*    scale = (float*)d_ws;                       // 256 * 4 B
    _Float16* wpack = (_Float16*)((char*)d_ws + 1024);    // 9*256*256 * 2 B, [p][o][c]

    hbc_scale_kernel<<<OUTC, 256, 0, stream>>>(w, scale);
    hbc_pack_kernel<<<(KPP * OUTC * CIN) / 256, 256, 0, stream>>>(w, wpack);

    dim3 grid(MTOT / MT, OUTC / OT);      // (784, 2)
    hbc_gemm_kernel<<<grid, 256, 0, stream>>>(x, wpack, scale, out);
}
